// LogNeuralCDE_13314398618438
// MI455X (gfx1250) — matrix operand order repaired and checked by reading, not compiled, not run on hardware
//
#include <hip/hip_runtime.h>
#include <hip/hip_bf16.h>
#include <math.h>

// ---------------------------------------------------------------------------
// Log-ODE Neural CDE forward, MI455X (gfx1250), wave32 + WMMA bf16 16x16x32.
// Persistent workgroups: 16 WGs x 256 threads, each WG integrates a 16-batch
// tile over 512 Heun steps. All GEMMs via v_wmma_f32_16x16x32_bf16.
// Weights forced to address_space(1) -> per-element global loads (LOADcnt only);
// activations live in LDS -> ds_load/ds_store/ds_add_f32 (DScnt only).
// ---------------------------------------------------------------------------

#define GLOBAL_AS __attribute__((address_space(1)))

typedef __attribute__((ext_vector_type(8)))  __bf16 v8bf;
typedef __attribute__((ext_vector_type(16))) __bf16 v16bf;
typedef __attribute__((ext_vector_type(8)))  float  v8f;

typedef const GLOBAL_AS __bf16* gbf16p;
typedef const GLOBAL_AS float*  gf32p;

#define HID    128
#define WD     6
#define VFH    256
#define NOUT   (WD * HID)     // 768
#define NINT   64
#define NSTEPS 512
#define BATCH  256
#define BT     16             // batch tile per workgroup
#define LSD    22             // logsig dim
#define NWAVE  8              // 256 threads / wave32

// ---- dynamic LDS layout (bytes) -------------------------------------------
#define OFF_XBF   0                         // bf16 y  (16 x 128)          4096
#define OFF_H0    4096                      // bf16 h0 (16 x 256)          8192
#define OFF_H1    12288                     // bf16 h1                     8192
#define OFF_H2    20480                     // bf16 h2                     8192
#define OFF_Z     28672                     // bf16 tanh out (16 x 768)   24576
#define OFF_U0    53248                     // bf16 tangent act (96x256)  49152
#define OFF_U1    102400                    // bf16 tangent act (96x256)  49152
#define OFF_Y     151552                    // f32 y (16 x 128)            8192
#define OFF_K1    159744                    // f32 k1                      8192
#define OFF_KACC  167936                    // f32 func accumulator        8192
#define OFF_CW    176128                    // f32 Cw (16 x 6)              384
#define OFF_CL    176512                    // f32 Cl (16 x 36)            2304
#define SMEM_BYTES 178816

// ---- fragment helpers ------------------------------------------------------
// A (16x32 bf16, MxK): lane&15 = M row; lane>>4 selects K blocks {hi*8..}, {16+hi*8..}
static __device__ __forceinline__ v16bf frag2(const __bf16* p0, const __bf16* p1) {
  v8bf lo = *(const v8bf*)p0;
  v8bf hi = *(const v8bf*)p1;
  return __builtin_shufflevector(lo, hi, 0,1,2,3,4,5,6,7,8,9,10,11,12,13,14,15);
}
// loads 16 contiguous bf16 from p; unused (mm_tile gathers B: lane&15 = N col, elements = K)
static __device__ __forceinline__ v16bf fragB(gbf16p p) {
  return *(const GLOBAL_AS v16bf*)p;
}
static __device__ __forceinline__ v8f splat8(float v) {
  v8f r = {v, v, v, v, v, v, v, v};
  return r;
}

// One 16x16 output tile, K = KD contraction.
// Arow: LDS activation base already offset to (lane&15) row + column offset.
// Bt:   K-major bf16 weights (KD rows x ldB cols) in global memory.
template <int KD>
static __device__ __forceinline__ v8f
mm_tile(const __bf16* Arow, int ldA, gbf16p Bt, int ldB, int nBase,
        v8f acc, int lane) {
  const int hi = lane >> 4;
#pragma unroll
  for (int kb = 0; kb < KD / 32; ++kb) {
    v16bf a = frag2(Arow + kb * 32 + hi * 8, Arow + kb * 32 + 16 + hi * 8);
    v16bf b;
#pragma unroll
    for (int s = 0; s < 16; ++s)
      b[s] = Bt[(kb * 32 + (s < 8 ? hi * 8 + s : 16 + hi * 8 + (s - 8))) * ldB + nBase + (lane & 15)];
    acc = __builtin_amdgcn_wmma_f32_16x16x32_bf16(false, a, false, b,
                                                  (short)0, acc, false, false);
  }
  return acc;
}

static __device__ __forceinline__ void lds_atomic_add(float* p, float v) {
  __hip_atomic_fetch_add(p, v, __ATOMIC_RELAXED, __HIP_MEMORY_SCOPE_WORKGROUP);
}

// ---- one evaluation of func(t, y): result accumulated (x NINT) into kacc ---
static __device__ __forceinline__ void
func_eval(char* sm, int idx, gf32p logsig, int b0,
          gbf16p W0t, gbf16p W1t, gbf16p W2t, gbf16p Wft,
          gf32p vb0, gf32p vb1, gf32p vb2, gf32p vbf) {
  __bf16* Xbf = (__bf16*)(sm + OFF_XBF);
  __bf16* H0  = (__bf16*)(sm + OFF_H0);
  __bf16* H1  = (__bf16*)(sm + OFF_H1);
  __bf16* H2  = (__bf16*)(sm + OFF_H2);
  __bf16* Z   = (__bf16*)(sm + OFF_Z);
  __bf16* U0  = (__bf16*)(sm + OFF_U0);
  __bf16* U1  = (__bf16*)(sm + OFF_U1);
  float*  kacc = (float*)(sm + OFF_KACC);
  float*  Cw   = (float*)(sm + OFF_CW);
  float*  Cl   = (float*)(sm + OFF_CL);

  const int tid  = threadIdx.x;
  const int lane = tid & 31;
  const int wid  = tid >> 5;
  const int nh   = lane & 15;
  const int hi   = lane >> 4;

  // Zero func accumulator; build per-batch coefficient tables from logsig.
  for (int e = tid; e < BT * HID; e += blockDim.x) kacc[e] = 0.f;
  if (tid < BT) {
    gf32p ls = logsig + (size_t)(b0 + tid) * NINT * LSD + (size_t)(idx - 1) * LSD;
    const float s = (float)NINT;   // 1 / interval width
#pragma unroll
    for (int w = 0; w < WD; ++w) Cw[tid * WD + w] = s * ls[1 + w];
    float* cl = Cl + tid * 36;
#pragma unroll
    for (int q = 0; q < 36; ++q) cl[q] = 0.f;
    int p = 0;
    for (int i = 0; i < WD; ++i)
      for (int j = i + 1; j < WD; ++j, ++p) {
        float v = s * ls[1 + WD + p];
        cl[j * WD + i] += v;   // + jvps[j][i]
        cl[i * WD + j] -= v;   // - jvps[i][j]
      }
  }
  __syncthreads();

  // ---- primal layer 0: (16x128) x W0t -> relu -> H0 (16x256) ----
  for (int tile = wid; tile < VFH / 16; tile += NWAVE) {
    const int nBase = tile * 16, n = nBase + nh;
    v8f acc = splat8(vb0[n]);
    acc = mm_tile<HID>(Xbf + nh * HID, HID, W0t, VFH, nBase, acc, lane);
#pragma unroll
    for (int r = 0; r < 8; ++r) {
      const int m = r + 8 * hi;
      float v = acc[r];
      H0[m * VFH + n] = (__bf16)(v > 0.f ? v : 0.f);
    }
  }
  __syncthreads();

  // ---- primal layer 1 ----
  for (int tile = wid; tile < VFH / 16; tile += NWAVE) {
    const int nBase = tile * 16, n = nBase + nh;
    v8f acc = splat8(vb1[n]);
    acc = mm_tile<VFH>(H0 + nh * VFH, VFH, W1t, VFH, nBase, acc, lane);
#pragma unroll
    for (int r = 0; r < 8; ++r) {
      const int m = r + 8 * hi;
      float v = acc[r];
      H1[m * VFH + n] = (__bf16)(v > 0.f ? v : 0.f);
    }
  }
  __syncthreads();

  // ---- primal layer 2 ----
  for (int tile = wid; tile < VFH / 16; tile += NWAVE) {
    const int nBase = tile * 16, n = nBase + nh;
    v8f acc = splat8(vb2[n]);
    acc = mm_tile<VFH>(H1 + nh * VFH, VFH, W2t, VFH, nBase, acc, lane);
#pragma unroll
    for (int r = 0; r < 8; ++r) {
      const int m = r + 8 * hi;
      float v = acc[r];
      H2[m * VFH + n] = (__bf16)(v > 0.f ? v : 0.f);
    }
  }
  __syncthreads();

  // ---- primal final: tanh, store Z, accumulate ls1 . vfo into kacc ----
  for (int tile = wid; tile < NOUT / 16; tile += NWAVE) {
    const int nBase = tile * 16, n = nBase + nh;
    const int w = n >> 7, h = n & 127;
    v8f acc = splat8(vbf[n]);
    acc = mm_tile<VFH>(H2 + nh * VFH, VFH, Wft, NOUT, nBase, acc, lane);
#pragma unroll
    for (int r = 0; r < 8; ++r) {
      const int m = r + 8 * hi;
      float z = tanhf(acc[r]);
      Z[m * NOUT + n] = (__bf16)z;
      lds_atomic_add(&kacc[m * HID + h], Cw[m * WD + w] * z);
    }
  }
  __syncthreads();

  // ---- tangent layer 0: seeds are columns of vfo (Z blocks), mask by H0 ----
  for (int job = wid; job < WD * (VFH / 16); job += NWAVE) {
    const int a = job / (VFH / 16), tile = job % (VFH / 16);
    const int nBase = tile * 16, n = nBase + nh;
    v8f acc = splat8(0.f);
    acc = mm_tile<HID>(Z + nh * NOUT + a * HID, NOUT, W0t, VFH, nBase, acc, lane);
#pragma unroll
    for (int r = 0; r < 8; ++r) {
      const int m = r + 8 * hi;
      float msk = (float)H0[m * VFH + n];
      U0[(a * BT + m) * VFH + n] = msk > 0.f ? (__bf16)acc[r] : (__bf16)0.f;
    }
  }
  __syncthreads();

  // ---- tangent layer 1: mask by H1 ----
  for (int job = wid; job < WD * (VFH / 16); job += NWAVE) {
    const int a = job / (VFH / 16), tile = job % (VFH / 16);
    const int nBase = tile * 16, n = nBase + nh;
    v8f acc = splat8(0.f);
    acc = mm_tile<VFH>(U0 + (a * BT + nh) * VFH, VFH, W1t, VFH, nBase, acc, lane);
#pragma unroll
    for (int r = 0; r < 8; ++r) {
      const int m = r + 8 * hi;
      float msk = (float)H1[m * VFH + n];
      U1[(a * BT + m) * VFH + n] = msk > 0.f ? (__bf16)acc[r] : (__bf16)0.f;
    }
  }
  __syncthreads();

  // ---- tangent layer 2: mask by H2 ----
  for (int job = wid; job < WD * (VFH / 16); job += NWAVE) {
    const int a = job / (VFH / 16), tile = job % (VFH / 16);
    const int nBase = tile * 16, n = nBase + nh;
    v8f acc = splat8(0.f);
    acc = mm_tile<VFH>(U1 + (a * BT + nh) * VFH, VFH, W2t, VFH, nBase, acc, lane);
#pragma unroll
    for (int r = 0; r < 8; ++r) {
      const int m = r + 8 * hi;
      float msk = (float)H2[m * VFH + n];
      U0[(a * BT + m) * VFH + n] = msk > 0.f ? (__bf16)acc[r] : (__bf16)0.f;
    }
  }
  __syncthreads();

  // ---- tangent final: scale by tanh' = 1 - z^2, contract with Cl into kacc ----
  for (int job = wid; job < WD * (NOUT / 16); job += NWAVE) {
    const int a = job / (NOUT / 16), tile = job % (NOUT / 16);
    const int nBase = tile * 16, n = nBase + nh;
    const int w = n >> 7, h = n & 127;
    v8f acc = splat8(0.f);
    acc = mm_tile<VFH>(U0 + (a * BT + nh) * VFH, VFH, Wft, NOUT, nBase, acc, lane);
#pragma unroll
    for (int r = 0; r < 8; ++r) {
      const int m = r + 8 * hi;
      float z = (float)Z[m * NOUT + n];
      float val = (1.f - z * z) * acc[r];
      lds_atomic_add(&kacc[m * HID + h], Cl[m * 36 + a * WD + w] * val);
    }
  }
  __syncthreads();
}

// ---- main persistent kernel ------------------------------------------------
__global__ __launch_bounds__(256, 1)
void logcde_kernel(const float* __restrict__ logsig_, const float* __restrict__ x0_,
                   const __bf16* __restrict__ W0_, const __bf16* __restrict__ W1_,
                   const __bf16* __restrict__ W2_, const __bf16* __restrict__ Wf_,
                   const float* __restrict__ vb0_, const float* __restrict__ vb1_,
                   const float* __restrict__ vb2_, const float* __restrict__ vbf_,
                   const float* __restrict__ lin1W_, const float* __restrict__ lin1b_,
                   const float* __restrict__ lin2W_, const float* __restrict__ lin2b_,
                   float* __restrict__ out_) {
  // Force global address space on all bulk-traffic pointers.
  gf32p  logsig = (gf32p)logsig_;
  gf32p  x0     = (gf32p)x0_;
  gbf16p W0t    = (gbf16p)W0_;
  gbf16p W1t    = (gbf16p)W1_;
  gbf16p W2t    = (gbf16p)W2_;
  gbf16p Wft    = (gbf16p)Wf_;
  gf32p  vb0    = (gf32p)vb0_;
  gf32p  vb1    = (gf32p)vb1_;
  gf32p  vb2    = (gf32p)vb2_;
  gf32p  vbf    = (gf32p)vbf_;
  gf32p  lin1W  = (gf32p)lin1W_;
  gf32p  lin1b  = (gf32p)lin1b_;
  gf32p  lin2W  = (gf32p)lin2W_;
  gf32p  lin2b  = (gf32p)lin2b_;
  GLOBAL_AS float* out = (GLOBAL_AS float*)out_;

  extern __shared__ __align__(32) char sm[];
  __bf16* Xbf  = (__bf16*)(sm + OFF_XBF);
  float*  y    = (float*)(sm + OFF_Y);
  float*  k1   = (float*)(sm + OFF_K1);
  float*  kacc = (float*)(sm + OFF_KACC);

  const int tid = threadIdx.x;
  const int b0  = blockIdx.x * BT;

  // y0 = lin1_W @ x0 + lin1_b
  for (int e = tid; e < BT * HID; e += blockDim.x) {
    const int m = e >> 7, h = e & 127;
    float acc = lin1b[h];
#pragma unroll
    for (int w = 0; w < WD; ++w)
      acc += lin1W[h * WD + w] * x0[(b0 + m) * WD + w];
    y[e] = acc;
    Xbf[e] = (__bf16)acc;
  }
  __syncthreads();

  const float dt = 1.0f / (float)NSTEPS;
  for (int n = 0; n < NSTEPS; ++n) {
    int idx1 = (n + 7) >> 3;  if (idx1 < 1) idx1 = 1;          // searchsorted(t_n)
    func_eval(sm, idx1, logsig, b0, W0t, W1t, W2t, Wft, vb0, vb1, vb2, vbf);

    for (int e = tid; e < BT * HID; e += blockDim.x) {          // k1; y + dt*k1
      float kv = kacc[e];
      k1[e] = kv;
      Xbf[e] = (__bf16)(y[e] + dt * kv);
    }
    __syncthreads();

    int idx2 = (n + 8) >> 3;  if (idx2 > NINT) idx2 = NINT;     // searchsorted(t_n + dt)
    func_eval(sm, idx2, logsig, b0, W0t, W1t, W2t, Wft, vb0, vb1, vb2, vbf);

    for (int e = tid; e < BT * HID; e += blockDim.x) {          // Heun update
      float yn = y[e] + 0.5f * dt * (k1[e] + kacc[e]);
      y[e] = yn;
      Xbf[e] = (__bf16)yn;
    }
    __syncthreads();
  }

  // classification head: softmax(lin2_W @ yT + lin2_b)
  float* logits = kacc;  // reuse
  for (int e = tid; e < BT * 10; e += blockDim.x) {
    const int m = e / 10, c = e % 10;
    float acc = lin2b[c];
    for (int h = 0; h < HID; ++h) acc += lin2W[c * HID + h] * y[m * HID + h];
    logits[e] = acc;
  }
  __syncthreads();
  if (tid < BT) {
    const int m = tid;
    float mx = -1e30f;
    for (int c = 0; c < 10; ++c) mx = fmaxf(mx, logits[m * 10 + c]);
    float ex[10], sum = 0.f;
    for (int c = 0; c < 10; ++c) { ex[c] = expf(logits[m * 10 + c] - mx); sum += ex[c]; }
    float inv = 1.f / sum;
    for (int c = 0; c < 10; ++c) out[(b0 + m) * 10 + c] = ex[c] * inv;
  }
}

// f32 (N,K) row-major -> bf16 (K,N) K-major (B-fragment friendly)
__global__ void convert_kt_bf16(const float* __restrict__ W, __bf16* __restrict__ Wt,
                                int N, int K) {
  int idx = blockIdx.x * blockDim.x + threadIdx.x;
  if (idx >= N * K) return;
  int n = idx / K, k = idx % K;
  ((GLOBAL_AS __bf16*)Wt)[k * N + n] = (__bf16)((gf32p)W)[idx];
}

extern "C" void kernel_launch(void* const* d_in, const int* in_sizes, int n_in,
                              void* d_out, int out_size, void* d_ws, size_t ws_size,
                              hipStream_t stream) {
  (void)in_sizes; (void)n_in; (void)out_size; (void)ws_size;
  // input order: ts, intervals, logsig, x0, vf_W0, vf_b0, vf_W1, vf_b1,
  //              vf_W2, vf_b2, vf_Wf, vf_bf, lin1_W, lin1_b, lin2_W, lin2_b
  const float* logsig = (const float*)d_in[2];
  const float* x0     = (const float*)d_in[3];
  const float* W0     = (const float*)d_in[4];
  const float* b0v    = (const float*)d_in[5];
  const float* W1     = (const float*)d_in[6];
  const float* b1v    = (const float*)d_in[7];
  const float* W2     = (const float*)d_in[8];
  const float* b2v    = (const float*)d_in[9];
  const float* Wf     = (const float*)d_in[10];
  const float* bfv    = (const float*)d_in[11];
  const float* lin1W  = (const float*)d_in[12];
  const float* lin1b  = (const float*)d_in[13];
  const float* lin2W  = (const float*)d_in[14];
  const float* lin2b  = (const float*)d_in[15];

  __bf16* W0t = (__bf16*)d_ws;                 // (128, 256)
  __bf16* W1t = W0t + VFH * HID;               // (256, 256)
  __bf16* W2t = W1t + VFH * VFH;               // (256, 256)
  __bf16* Wft = W2t + VFH * VFH;               // (256, 768)

  convert_kt_bf16<<<(VFH * HID + 255) / 256, 256, 0, stream>>>(W0, W0t, VFH, HID);
  convert_kt_bf16<<<(VFH * VFH + 255) / 256, 256, 0, stream>>>(W1, W1t, VFH, VFH);
  convert_kt_bf16<<<(VFH * VFH + 255) / 256, 256, 0, stream>>>(W2, W2t, VFH, VFH);
  convert_kt_bf16<<<(NOUT * VFH + 255) / 256, 256, 0, stream>>>(Wf, Wft, NOUT, VFH);

  hipFuncSetAttribute((const void*)logcde_kernel,
                      hipFuncAttributeMaxDynamicSharedMemorySize, SMEM_BYTES);
  logcde_kernel<<<BATCH / BT, 256, SMEM_BYTES, stream>>>(
      logsig, x0, W0t, W1t, W2t, Wft, b0v, b1v, b2v, bfv,
      lin1W, lin1b, lin2W, lin2b, (float*)d_out);
}
